// Gat_bayes_11295763988536
// MI455X (gfx1250) — matrix operand order repaired and checked by reading, not compiled, not run on hardware
//
#include <hip/hip_runtime.h>
#include <hip/hip_bf16.h>
#include <math.h>

typedef __attribute__((ext_vector_type(16))) _Float16 v16h;
typedef __attribute__((ext_vector_type(8)))  _Float16 v8h;
typedef __attribute__((ext_vector_type(8)))  float    v8f;

#define NNODE 13627
#define NEDGE 504378
#define DFEAT 500

// ---------- helpers: order-preserving float<->uint for atomic max ----------
__device__ __forceinline__ unsigned f2ord(float f) {
  unsigned u = __float_as_uint(f);
  return (u & 0x80000000u) ? ~u : (u | 0x80000000u);
}
__device__ __forceinline__ float ord2f(unsigned e) {
  unsigned u = (e & 0x80000000u) ? (e ^ 0x80000000u) : ~e;
  return __uint_as_float(u);
}

// ---------- zero fill (grid-stride) ----------
__global__ void k_zero(float* p, long long n) {
  long long i = (long long)blockIdx.x * blockDim.x + threadIdx.x;
  long long stride = (long long)gridDim.x * blockDim.x;
  for (; i < n; i += stride) p[i] = 0.0f;
}

// ---------- f32 [M,K] -> f16 padded [Mp,Kp], zero fill ----------
__global__ void k_cvt_pad(const float* __restrict__ A, _Float16* __restrict__ Ah,
                          int M, int K, int Mp, int Kp) {
  long long total = (long long)Mp * Kp;
  long long i = (long long)blockIdx.x * blockDim.x + threadIdx.x;
  if (i >= total) return;
  int row = (int)(i / Kp), k = (int)(i % Kp);
  float v = (row < M && k < K) ? A[(long long)row * K + k] : 0.0f;
  Ah[i] = (_Float16)v;
}

// ---------- f32 W[K,Nc] -> f16 transposed padded Wt[Ncp,Kp], zero fill ------
__global__ void k_cvt_pad_t(const float* __restrict__ W, _Float16* __restrict__ Wt,
                            int K, int Nc, int Ncp, int Kp) {
  long long total = (long long)Ncp * Kp;
  long long i = (long long)blockIdx.x * blockDim.x + threadIdx.x;
  if (i >= total) return;
  int n = (int)(i / Kp), k = (int)(i % Kp);
  float v = (n < Nc && k < K) ? W[(long long)k * Nc + n] : 0.0f;
  Wt[i] = (_Float16)v;
}

// ---------- WMMA GEMM on pre-converted/padded f16 operands ------------------
// Ah: [Mp, KP] f16 row-major (Mp = tilesM*16).  Bt: [Ncp, KP] f16 (col n of W
// stored as row n).  One wave per 16x16 C tile, KP fully unrolled.
// A lane layout (ISA 7.12.2): lane r=lane&15 row M=r; halves 0-7 -> K=kb+hi*8+0..7,
// halves 8-15 -> K=kb+16+hi*8+0..7  => two contiguous B128 loads.
// B lane layout: lane r col N=r; same K mapping as A => two contiguous B128 loads.
template <int KP>
__global__ void k_gemm_wmma(const _Float16* __restrict__ Ah, const _Float16* __restrict__ Bt,
                            const float* __restrict__ bias, float* __restrict__ C,
                            int M, int Nc, int relu, int tilesN) {
  int lane = threadIdx.x & 31;
  int wib  = threadIdx.x >> 5;
  int tilesM = (M + 15) >> 4;
  int tile = blockIdx.x * (blockDim.x >> 5) + wib;
  if (tile >= tilesM * tilesN) return;   // wave-uniform: EXEC stays all-ones
  int tM = tile / tilesN, tN = tile % tilesN;
  int r  = lane & 15;
  int hi = lane >> 4;
  const _Float16* aBase = Ah + (long long)(tM * 16 + r) * KP + hi * 8;
  const _Float16* bBase = Bt + (long long)(tN * 16 + r) * KP + hi * 8;

  v8f acc = {};
#pragma unroll
  for (int kb = 0; kb < KP; kb += 32) {
    v8h a0 = *(const v8h*)(aBase + kb);
    v8h a1 = *(const v8h*)(aBase + kb + 16);
    v8h b0 = *(const v8h*)(bBase + kb);
    v8h b1 = *(const v8h*)(bBase + kb + 16);
    v16h av = __builtin_shufflevector(a0, a1, 0,1,2,3,4,5,6,7,8,9,10,11,12,13,14,15);
    v16h bv = __builtin_shufflevector(b0, b1, 0,1,2,3,4,5,6,7,8,9,10,11,12,13,14,15);
    acc = __builtin_amdgcn_wmma_f32_16x16x32_f16(
        /*neg_a=*/false, av, /*neg_b=*/false, bv,
        /*c_mod=*/(short)0, acc, /*reuse_a=*/false, /*reuse_b=*/false);
  }
#pragma unroll
  for (int v = 0; v < 8; ++v) {
    int orow = tM * 16 + v + hi * 8;
    int ocol = tN * 16 + r;
    if (orow < M && ocol < Nc) {
      float val = acc[v];
      if (bias) val += bias[ocol];
      if (relu) val = fmaxf(val, 0.0f);
      C[(long long)orow * Nc + ocol] = val;
    }
  }
}

// ---------- edge attention scores + atomic segment max ----------
__global__ void k_edge_scores(const int* __restrict__ ei, int E, int E2,
                              const float* __restrict__ xl, const float* __restrict__ xr,
                              const float* __restrict__ att, int F,
                              float* __restrict__ e_out, unsigned* __restrict__ m_enc) {
  int wid  = (blockIdx.x * blockDim.x + threadIdx.x) >> 5;
  int lane = threadIdx.x & 31;
  if (wid >= E2) return;
  int s, d;
  if (wid < E) { s = ei[wid]; d = ei[E + wid]; } else { s = wid - E; d = s; }
  float acc = 0.0f;
  for (int f = lane; f < F; f += 32) {
    float v = xl[(long long)s * F + f] + xr[(long long)d * F + f];
    v = (v > 0.0f) ? v : 0.2f * v;
    acc += v * att[f];
  }
#pragma unroll
  for (int off = 16; off > 0; off >>= 1) acc += __shfl_xor(acc, off, 32);
  if (lane == 0) {
    e_out[wid] = acc;
    atomicMax(&m_enc[d], f2ord(acc));
  }
}

// ---------- exp(e - max[dst]) and segment sum ----------
__global__ void k_edge_exp(const int* __restrict__ ei, int E, int E2,
                           float* __restrict__ e_io, const unsigned* __restrict__ m_enc,
                           float* __restrict__ s_sum) {
  int i = blockIdx.x * blockDim.x + threadIdx.x;
  if (i >= E2) return;
  int d = (i < E) ? ei[E + i] : (i - E);
  float m = ord2f(m_enc[d]);
  float ex = expf(e_io[i] - m);
  e_io[i] = ex;
  atomicAdd(&s_sum[d], ex);
}

// ---------- weighted scatter: out[dst] += a * xl[src] ----------
__global__ void k_edge_scatter(const int* __restrict__ ei, int E, int E2,
                               const float* __restrict__ e_ex, const float* __restrict__ s_sum,
                               const float* __restrict__ xl, int F, float* __restrict__ out) {
  int wid  = (blockIdx.x * blockDim.x + threadIdx.x) >> 5;
  int lane = threadIdx.x & 31;
  if (wid >= E2) return;
  int s, d;
  if (wid < E) { s = ei[wid]; d = ei[E + wid]; } else { s = wid - E; d = s; }
  float a = e_ex[wid] / (s_sum[d] + 1e-16f);
  for (int f = lane; f < F; f += 32)
    atomicAdd(&out[(long long)d * F + f], a * xl[(long long)s * F + f]);
}

// ---------- elementwise bias (+optional relu), in place ----------
__global__ void k_bias_act(float* __restrict__ p, const float* __restrict__ bias,
                           int Nc, long long total, int relu) {
  long long i = (long long)blockIdx.x * blockDim.x + threadIdx.x;
  if (i >= total) return;
  float v = p[i] + bias[(int)(i % Nc)];
  if (relu) v = fmaxf(v, 0.0f);
  p[i] = v;
}

// ---------- c = a + b ----------
__global__ void k_add(const float* __restrict__ a, const float* __restrict__ b,
                      float* __restrict__ c, long long n) {
  long long i = (long long)blockIdx.x * blockDim.x + threadIdx.x;
  if (i < n) c[i] = a[i] + b[i];
}

// ---------- per-node dot with a small weight vector (layer 3, F->1) ----------
__global__ void k_vecdot(const float* __restrict__ X, const float* __restrict__ w,
                         int F, int n, float* __restrict__ out) {
  int i = blockIdx.x * blockDim.x + threadIdx.x;
  if (i >= n) return;
  float a = 0.0f;
  for (int f = 0; f < F; ++f) a += X[(long long)i * F + f] * w[f];
  out[i] = a;
}

// ---------- link prediction loss accumulation (wave-per-edge dot) ----------
__global__ void k_loss(const int* __restrict__ ei, int E, const float* __restrict__ z,
                       int F, int neg, float* __restrict__ acc) {
  int wid  = (blockIdx.x * blockDim.x + threadIdx.x) >> 5;
  int lane = threadIdx.x & 31;
  if (wid >= E) return;
  int s = ei[wid], d = ei[E + wid];
  float dot = 0.0f;
  for (int f = lane; f < F; f += 32)
    dot += z[(long long)s * F + f] * z[(long long)d * F + f];
#pragma unroll
  for (int off = 16; off > 0; off >>= 1) dot += __shfl_xor(dot, off, 32);
  if (lane == 0) {
    float sig = 1.0f / (1.0f + expf(-dot));
    float l = neg ? -logf(1.0f - sig + 1e-15f) : -logf(sig + 1e-15f);
    atomicAdd(&acc[neg], l);
  }
}

// ---------- final scalars ----------
__global__ void k_finalize(float* __restrict__ out, const float* __restrict__ acc,
                           const float* __restrict__ c1, const float* __restrict__ c2,
                           int n, float invE) {
  if (blockIdx.x == 0 && threadIdx.x == 0) {
    out[n]     = acc[0] * invE + acc[1] * invE;
    out[n + 1] = c1[0];
    out[n + 2] = c2[0];
  }
}

extern "C" void kernel_launch(void* const* d_in, const int* in_sizes, int n_in,
                              void* d_out, int out_size, void* d_ws, size_t ws_size,
                              hipStream_t stream) {
  (void)in_sizes; (void)n_in; (void)out_size; (void)ws_size;
  const int N = NNODE, E = NEDGE, D = DFEAT;
  const int E2 = E + N;

  const float* x     = (const float*)d_in[0];
  const int*   ei    = (const int*)d_in[1];
  const int*   nei   = (const int*)d_in[2];
  const float* Wl1   = (const float*)d_in[3];
  const float* Wr1   = (const float*)d_in[4];
  const float* att1  = (const float*)d_in[5];
  const float* b1    = (const float*)d_in[6];
  const float* Wl2   = (const float*)d_in[7];
  const float* Wr2   = (const float*)d_in[8];
  const float* att2  = (const float*)d_in[9];
  const float* b2    = (const float*)d_in[10];
  const float* Wl3   = (const float*)d_in[11];
  const float* Wr3   = (const float*)d_in[12];
  const float* att3  = (const float*)d_in[13];
  const float* b3    = (const float*)d_in[14];
  const float* Wlin1 = (const float*)d_in[15];
  const float* blin1 = (const float*)d_in[16];
  const float* Wlin2 = (const float*)d_in[17];
  const float* blin2 = (const float*)d_in[18];
  const float* c1    = (const float*)d_in[19];
  const float* c2    = (const float*)d_in[20];
  float* out = (float*)d_out;

  // ------- padded shapes for WMMA -------
  const int MP   = ((N + 15) / 16) * 16;   // 13632
  const int KP5  = 512;                    // K=500 padded
  const int KP3  = 320;                    // K=300 padded
  const int TN3  = (300 + 15) / 16;        // 19 tiles -> Ncp 304
  const int TN1  = (100 + 15) / 16;        // 7 tiles  -> Ncp 112
  const int NCP3 = TN3 * 16, NCP1 = TN1 * 16;

  // ------- workspace layout: float region then f16 region -------
  const long long N300 = (long long)N * 300, N100 = (long long)N * 100;
  float* ws   = (float*)d_ws;
  float* bufA = ws;              // xl1 ; later: lin1 @bufA, lin2 @bufA+N100
  float* bufB = bufA + N300;     // xr1 ; later: xl3 @bufB, xr3 @bufB+N
  float* bufC = bufB + N300;     // out1 / h1
  float* bufP = bufC + N300;     // xl2 ; later: xo
  float* bufQ = bufP + N100;     // xr2 ; later: z
  float* bufH = bufQ + N100;     // out2 / h2
  float* ew   = bufH + N100;     // per-edge e / exp(e)  [E2]
  unsigned* m_enc = (unsigned*)(ew + E2);     // [N]
  float* s_sum = ew + E2 + N;                 // [N]
  float* scal  = s_sum + N;                   // [8] loss accumulators (padded)

  long long fEnd = 3 * N300 + 3 * N100 + (long long)E2 + 2LL * N + 8;
  fEnd = (fEnd + 7) & ~7LL;                   // 32B-align the f16 region
  _Float16* hbase = (_Float16*)(ws + fEnd);
  _Float16* xh    = hbase;                          // [MP,  KP5]
  _Float16* h1h   = xh    + (long long)MP * KP5;    // [MP,  KP3]
  _Float16* wl1t  = h1h   + (long long)MP * KP3;    // [NCP3,KP5]
  _Float16* wr1t  = wl1t  + (long long)NCP3 * KP5;
  _Float16* wl2t  = wr1t  + (long long)NCP3 * KP5;  // [NCP1,KP3]
  _Float16* wr2t  = wl2t  + (long long)NCP1 * KP3;
  _Float16* wn1t  = wr2t  + (long long)NCP1 * KP3;  // [NCP1,KP5]
  _Float16* wn2t  = wn1t  + (long long)NCP1 * KP5;

  const int TB = 256;
  auto cdiv = [](long long a, long long b) { return (int)((a + b - 1) / b); };
  const int eWaveBlk = cdiv((long long)E2 * 32, TB);
  const int eThrBlk  = cdiv(E2, TB);

  // ---------------- operand conversion (every call; deterministic) ----------
  k_cvt_pad<<<cdiv((long long)MP * KP5, TB), TB, 0, stream>>>(x, xh, N, D, MP, KP5);
  k_cvt_pad_t<<<cdiv((long long)NCP3 * KP5, TB), TB, 0, stream>>>(Wl1, wl1t, D, 300, NCP3, KP5);
  k_cvt_pad_t<<<cdiv((long long)NCP3 * KP5, TB), TB, 0, stream>>>(Wr1, wr1t, D, 300, NCP3, KP5);
  k_cvt_pad_t<<<cdiv((long long)NCP1 * KP3, TB), TB, 0, stream>>>(Wl2, wl2t, 300, 100, NCP1, KP3);
  k_cvt_pad_t<<<cdiv((long long)NCP1 * KP3, TB), TB, 0, stream>>>(Wr2, wr2t, 300, 100, NCP1, KP3);
  k_cvt_pad_t<<<cdiv((long long)NCP1 * KP5, TB), TB, 0, stream>>>(Wlin1, wn1t, D, 100, NCP1, KP5);
  k_cvt_pad_t<<<cdiv((long long)NCP1 * KP5, TB), TB, 0, stream>>>(Wlin2, wn2t, D, 100, NCP1, KP5);

  // ---------------- Layer 1 (F=300) ----------------
  k_zero<<<2048, TB, 0, stream>>>(bufC, N300);                    // out1 accumulator
  k_zero<<<64, TB, 0, stream>>>((float*)m_enc, 2LL * N + 8);      // max/sum/scal
  {
    int tiles  = ((N + 15) / 16) * TN3;
    int blocks = cdiv(tiles, TB / 32);
    k_gemm_wmma<KP5><<<blocks, TB, 0, stream>>>(xh, wl1t, nullptr, bufA, N, 300, 0, TN3);
    k_gemm_wmma<KP5><<<blocks, TB, 0, stream>>>(xh, wr1t, nullptr, bufB, N, 300, 0, TN3);
  }
  k_edge_scores<<<eWaveBlk, TB, 0, stream>>>(ei, E, E2, bufA, bufB, att1, 300, ew, m_enc);
  k_edge_exp<<<eThrBlk, TB, 0, stream>>>(ei, E, E2, ew, m_enc, s_sum);
  k_edge_scatter<<<eWaveBlk, TB, 0, stream>>>(ei, E, E2, ew, s_sum, bufA, 300, bufC);
  k_bias_act<<<cdiv(N300, TB), TB, 0, stream>>>(bufC, b1, 300, N300, 1);   // h1

  // ---------------- Layer 2 (F=100) ----------------
  k_cvt_pad<<<cdiv((long long)MP * KP3, TB), TB, 0, stream>>>(bufC, h1h, N, 300, MP, KP3);
  k_zero<<<2048, TB, 0, stream>>>(bufH, N100);
  k_zero<<<64, TB, 0, stream>>>((float*)m_enc, 2LL * N);
  {
    int tiles  = ((N + 15) / 16) * TN1;
    int blocks = cdiv(tiles, TB / 32);
    k_gemm_wmma<KP3><<<blocks, TB, 0, stream>>>(h1h, wl2t, nullptr, bufP, N, 100, 0, TN1);
    k_gemm_wmma<KP3><<<blocks, TB, 0, stream>>>(h1h, wr2t, nullptr, bufQ, N, 100, 0, TN1);
  }
  k_edge_scores<<<eWaveBlk, TB, 0, stream>>>(ei, E, E2, bufP, bufQ, att2, 100, ew, m_enc);
  k_edge_exp<<<eThrBlk, TB, 0, stream>>>(ei, E, E2, ew, m_enc, s_sum);
  k_edge_scatter<<<eWaveBlk, TB, 0, stream>>>(ei, E, E2, ew, s_sum, bufP, 100, bufH);
  k_bias_act<<<cdiv(N100, TB), TB, 0, stream>>>(bufH, b2, 100, N100, 1);   // h2

  // ---------------- Linear skip branches (fused bias+relu GEMM) -------------
  float* lin1 = bufA;            // xl1 is dead now
  float* lin2 = bufA + N100;
  {
    int tiles  = ((N + 15) / 16) * TN1;
    int blocks = cdiv(tiles, TB / 32);
    k_gemm_wmma<KP5><<<blocks, TB, 0, stream>>>(xh, wn1t, blin1, lin1, N, 100, 1, TN1);
    k_gemm_wmma<KP5><<<blocks, TB, 0, stream>>>(xh, wn2t, blin2, lin2, N, 100, 1, TN1);
  }
  float* xo = bufP;              // xl2 dead
  float* z  = bufQ;              // xr2 dead
  k_add<<<cdiv(N100, TB), TB, 0, stream>>>(bufH, lin1, xo, N100);
  k_add<<<cdiv(N100, TB), TB, 0, stream>>>(bufH, lin2, z, N100);

  // ---------------- Reconstruction loss ----------------
  int lWaveBlk = cdiv((long long)E * 32, TB);
  k_loss<<<lWaveBlk, TB, 0, stream>>>(ei,  E, z, 100, 0, scal);
  k_loss<<<lWaveBlk, TB, 0, stream>>>(nei, E, z, 100, 1, scal);

  // ---------------- Layer 3 (F=1) ----------------
  float* xl3 = bufB;             // xr1 dead
  float* xr3 = bufB + N;
  k_vecdot<<<cdiv(N, TB), TB, 0, stream>>>(xo, Wl3, 100, N, xl3);
  k_vecdot<<<cdiv(N, TB), TB, 0, stream>>>(xo, Wr3, 100, N, xr3);
  k_zero<<<64, TB, 0, stream>>>((float*)m_enc, 2LL * N);
  k_zero<<<64, TB, 0, stream>>>(out, N);
  k_edge_scores<<<eWaveBlk, TB, 0, stream>>>(ei, E, E2, xl3, xr3, att3, 1, ew, m_enc);
  k_edge_exp<<<eThrBlk, TB, 0, stream>>>(ei, E, E2, ew, m_enc, s_sum);
  k_edge_scatter<<<eWaveBlk, TB, 0, stream>>>(ei, E, E2, ew, s_sum, xl3, 1, out);
  k_bias_act<<<cdiv(N, TB), TB, 0, stream>>>(out, b3, 1, N, 0);
  k_finalize<<<1, 32, 0, stream>>>(out, scal, c1, c2, N, 1.0f / (float)E);
}
